// LlamaAttention_45380624450227
// MI455X (gfx1250) — compile-verified, matrix operand order checked by reading, not run on hardware
//
#include <hip/hip_runtime.h>
#include <hip/hip_bf16.h>
#include <stdint.h>

// Problem constants (from reference)
#define B_   2
#define S_   1024
#define P_   1024
#define E_   2048
#define H_   32
#define KV_  8
#define HD_  64
#define G_   4
#define CTX_ 2048
#define KVD_ 512   // KV*HD

typedef __bf16 bf16;
typedef __attribute__((ext_vector_type(16))) __bf16 bf16x16;
typedef __attribute__((ext_vector_type(8)))  __bf16 bf16x8;
typedef __attribute__((ext_vector_type(8)))  float  f32x8;

#define WMMA_BF16(A, Bv, C) \
  __builtin_amdgcn_wmma_f32_16x16x32_bf16(false, (A), false, (Bv), (short)0, (C), false, false)

static __device__ __forceinline__ bf16 f2bf(float f) {
  // round-to-nearest-even f32 -> bf16
  unsigned u = __float_as_uint(f);
  unsigned r = (u + 0x7FFFu + ((u >> 16) & 1u)) >> 16;
  unsigned short s = (unsigned short)r;
  return __builtin_bit_cast(bf16, s);
}

// Load a 16x32 bf16 WMMA A-fragment (or B-fragment: column n of B(KxN) ==
// row n of a matrix stored N-major x K) from row-major memory.
// CDNA5 16-bit striping: lanes 0-15 -> row=lane, K{kb..kb+7, kb+16..kb+23} with kb=0;
// lanes 16-31 -> row=lane-16, kb=8.  Two 16-byte loads per lane.
static __device__ __forceinline__ bf16x16 load_frag(const bf16* __restrict__ base, int ld) {
  const int lane = threadIdx.x & 31;
  const int row  = lane & 15;
  const int kb   = (lane < 16) ? 0 : 8;
  const bf16* p  = base + (size_t)row * ld + kb;
  bf16x8 lo = *(const bf16x8*)(p);
  bf16x8 hi = *(const bf16x8*)(p + 16);
  bf16x16 f;
#pragma unroll
  for (int i = 0; i < 8; ++i) { f[i] = lo[i]; f[i + 8] = hi[i]; }
  return f;
}

// ---------------------------------------------------------------- converts
__global__ void k_f32_to_bf16(const float* __restrict__ src, bf16* __restrict__ dst, int n) {
  int i = blockIdx.x * blockDim.x + threadIdx.x;
  if (i < n) dst[i] = f2bf(src[i]);
}

// ---------------------------------------------------------------- GEMM
// C[M][N] (f32) = A[M][K] (bf16, row-major) * B[N][K]^T (bf16, N-major).
// 4 waves/block; each wave owns a 64x64 tile (4x4 WMMA tiles, 16 WMMA / K-step,
// 32 flops per L2 byte) -> block covers 128x128.
__global__ void __launch_bounds__(128)
k_gemm_bf16_nt(const bf16* __restrict__ A, const bf16* __restrict__ Bm,
               float* __restrict__ C, int M, int N, int K) {
  const int wave = threadIdx.x >> 5;
  const int row0 = blockIdx.y * 128 + (wave >> 1) * 64;
  const int col0 = blockIdx.x * 128 + (wave & 1) * 64;

  f32x8 acc[4][4];
#pragma unroll
  for (int i = 0; i < 4; ++i)
#pragma unroll
    for (int j = 0; j < 4; ++j)
#pragma unroll
      for (int e = 0; e < 8; ++e) acc[i][j][e] = 0.f;

  const bf16* ap[4];
  const bf16* bp[4];
#pragma unroll
  for (int i = 0; i < 4; ++i) {
    ap[i] = A  + (size_t)(row0 + 16 * i) * K;
    bp[i] = Bm + (size_t)(col0 + 16 * i) * K;
  }

#pragma unroll 1
  for (int k = 0; k < K; k += 32) {
    bf16x16 af[4], bfr[4];
#pragma unroll
    for (int i = 0; i < 4; ++i) af[i]  = load_frag(ap[i] + k, K);
#pragma unroll
    for (int i = 0; i < 4; ++i) bfr[i] = load_frag(bp[i] + k, K);
    if (k + 256 < K) {             // pull next tiles toward L2/L0
      __builtin_prefetch(ap[0] + k + 256, 0, 1);
      __builtin_prefetch(bp[0] + k + 256, 0, 1);
    }
#pragma unroll
    for (int i = 0; i < 4; ++i)
#pragma unroll
      for (int j = 0; j < 4; ++j)
        acc[i][j] = WMMA_BF16(af[i], bfr[j], acc[i][j]);
  }

  const int lane = threadIdx.x & 31;
  const int n  = lane & 15;
  const int rb = (lane < 16) ? 0 : 8;   // C/D layout: elem j -> row j+rb, col n
#pragma unroll
  for (int i = 0; i < 4; ++i)
#pragma unroll
    for (int j = 0; j < 4; ++j)
#pragma unroll
      for (int e = 0; e < 8; ++e)
        C[(size_t)(row0 + 16 * i + rb + e) * N + col0 + 16 * j + n] = acc[i][j][e];
}

// ---------------------------------------------------------------- RoPE / KV staging
__global__ void k_rope_q(const float* __restrict__ qf, const float* __restrict__ cosb,
                         const float* __restrict__ sinb, bf16* __restrict__ qb) {
  int i = blockIdx.x * blockDim.x + threadIdx.x;
  if (i >= B_ * H_ * S_ * HD_) return;
  int hd = i & 63;
  int s  = (i >> 6) & (S_ - 1);
  int h  = (i >> 16) & (H_ - 1);
  int b  = i >> 21;
  size_t row = (size_t)(b * S_ + s) * E_;
  float v       = qf[row + h * HD_ + hd];
  float partner = qf[row + h * HD_ + (hd ^ 32)];
  float rot = (hd < 32) ? -partner : partner;
  size_t ci = (size_t)(b * S_ + s) * HD_ + hd;
  float o = v * cosb[ci] + rot * sinb[ci];
  qb[((size_t)(b * H_ + h) * S_ + s) * HD_ + hd] = f2bf(o);
}

__global__ void k_build_keys(const float* __restrict__ kf, const float* __restrict__ cache_k,
                             const float* __restrict__ cosb, const float* __restrict__ sinb,
                             bf16* __restrict__ keysb) {
  int i = blockIdx.x * blockDim.x + threadIdx.x;
  if (i >= B_ * KV_ * CTX_ * HD_) return;
  int hd = i & 63;
  int t  = (i >> 6) & (CTX_ - 1);
  int kv = (i >> 17) & (KV_ - 1);
  int b  = i >> 20;
  float val;
  if (t < P_) {
    val = cache_k[(size_t)i];                       // (B,KV,CTX,HD) flat == i
  } else {
    int s = t - P_;
    size_t row = (size_t)(b * S_ + s) * KVD_;
    float v       = kf[row + kv * HD_ + hd];
    float partner = kf[row + kv * HD_ + (hd ^ 32)];
    float rot = (hd < 32) ? -partner : partner;
    size_t ci = (size_t)(b * S_ + s) * HD_ + hd;
    val = v * cosb[ci] + rot * sinb[ci];
  }
  keysb[i] = f2bf(val);
}

__global__ void k_build_valsT(const float* __restrict__ vf, const float* __restrict__ cache_v,
                              bf16* __restrict__ valst) {
  int i = blockIdx.x * blockDim.x + threadIdx.x;
  if (i >= B_ * KV_ * HD_ * CTX_) return;
  int t  = i & (CTX_ - 1);
  int hd = (i >> 11) & 63;
  int kv = (i >> 17) & (KV_ - 1);
  int b  = i >> 20;
  float val;
  if (t < P_) {
    val = cache_v[((size_t)(b * KV_ + kv) * CTX_ + t) * HD_ + hd];
  } else {
    val = vf[(size_t)(b * S_ + (t - P_)) * KVD_ + kv * HD_ + hd];
  }
  valst[i] = f2bf(val);                              // (B,KV,HD,CTX) flat == i
}

// ---------------------------------------------------------------- flash attention
// One wave per (b, h, 16-query tile); streams 32-key chunks with online softmax.
__global__ void __launch_bounds__(32)
k_flash_attn(const bf16* __restrict__ qb, const bf16* __restrict__ keysb,
             const bf16* __restrict__ valst, float* __restrict__ attnf) {
  const int qt = blockIdx.x;
  const int h  = blockIdx.y;
  const int b  = blockIdx.z;
  const int kv = h / G_;
  const int lane = threadIdx.x;
  const int n  = lane & 15;
  const int rb = (lane < 16) ? 0 : 8;

  const bf16* Q  = qb    + ((size_t)(b * H_ + h) * S_ + qt * 16) * HD_;
  const bf16* Kp = keysb + (size_t)(b * KV_ + kv) * CTX_ * HD_;
  const bf16* Vt = valst + (size_t)(b * KV_ + kv) * HD_ * CTX_;

  const bf16x16 qa0 = load_frag(Q,      HD_);   // K = 0..31 of HD
  const bf16x16 qa1 = load_frag(Q + 32, HD_);   // K = 32..63

  float m[8], l[8];
  f32x8 acc[4];
#pragma unroll
  for (int j = 0; j < 8; ++j) { m[j] = -1e30f; l[j] = 0.f; }
#pragma unroll
  for (int hh = 0; hh < 4; ++hh)
#pragma unroll
    for (int j = 0; j < 8; ++j) acc[hh][j] = 0.f;

  __shared__ __align__(16) bf16 pt[16][40];     // 80B row stride, 16B aligned slices

  const int qposBase = P_ + qt * 16;            // query position of tile row 0
  const int nchunk   = (qposBase + 15 + 32) / 32;

  for (int c = 0; c < nchunk; ++c) {
    const int t0 = c * 32;

    // ---- scores: two 16x16 tiles over 32 keys, K-depth = HD = 64
    f32x8 s0, s1;
#pragma unroll
    for (int j = 0; j < 8; ++j) { s0[j] = 0.f; s1[j] = 0.f; }
    {
      bf16x16 kb0 = load_frag(Kp + (size_t)t0 * HD_,            HD_);
      bf16x16 kb1 = load_frag(Kp + (size_t)t0 * HD_ + 32,       HD_);
      s0 = WMMA_BF16(qa0, kb0, s0);
      s0 = WMMA_BF16(qa1, kb1, s0);
      bf16x16 kb2 = load_frag(Kp + (size_t)(t0 + 16) * HD_,      HD_);
      bf16x16 kb3 = load_frag(Kp + (size_t)(t0 + 16) * HD_ + 32, HD_);
      s1 = WMMA_BF16(qa0, kb2, s1);
      s1 = WMMA_BF16(qa1, kb3, s1);
    }

    // ---- scale + causal mask (analytic equivalent of the -1e9 mask)
#pragma unroll
    for (int j = 0; j < 8; ++j) {
      const int qpos = qposBase + rb + j;
      s0[j] = (t0 + n      <= qpos) ? s0[j] * 0.125f : -1e30f;   // 1/sqrt(64)
      s1[j] = (t0 + 16 + n <= qpos) ? s1[j] * 0.125f : -1e30f;
    }

    // ---- online softmax (row M=j+rb lives in elem j across a 16-lane half)
#pragma unroll
    for (int j = 0; j < 8; ++j) {
      float rmax = fmaxf(s0[j], s1[j]);
#pragma unroll
      for (int off = 1; off < 16; off <<= 1)
        rmax = fmaxf(rmax, __shfl_xor(rmax, off, 32));
      const float nm   = fmaxf(m[j], rmax);
      const float corr = __expf(m[j] - nm);
      const float p0   = __expf(s0[j] - nm);
      const float p1   = __expf(s1[j] - nm);
      float rsum = p0 + p1;
#pragma unroll
      for (int off = 1; off < 16; off <<= 1)
        rsum += __shfl_xor(rsum, off, 32);
      l[j] = l[j] * corr + rsum;
      m[j] = nm;
#pragma unroll
      for (int hh = 0; hh < 4; ++hh) acc[hh][j] *= corr;
      s0[j] = p0;
      s1[j] = p1;
    }

    // ---- transpose P (16x32) through LDS into an A-fragment
#pragma unroll
    for (int j = 0; j < 8; ++j) {
      pt[rb + j][n]      = f2bf(s0[j]);
      pt[rb + j][16 + n] = f2bf(s1[j]);
    }
    asm volatile("s_wait_dscnt 0x0" ::: "memory");
    bf16x16 pa;
    {
      const int prow = lane & 15;
      const int kbo  = (lane < 16) ? 0 : 8;
      bf16x8 lo = *(const bf16x8*)(&pt[prow][kbo]);
      bf16x8 hi = *(const bf16x8*)(&pt[prow][kbo + 16]);
#pragma unroll
      for (int i2 = 0; i2 < 8; ++i2) { pa[i2] = lo[i2]; pa[i2 + 8] = hi[i2]; }
    }
    asm volatile("s_wait_dscnt 0x0" ::: "memory");

    // ---- acc += P * V   (V^T stored HD-major -> contiguous B-fragment rows)
#pragma unroll
    for (int hh = 0; hh < 4; ++hh) {
      bf16x16 vb = load_frag(Vt + (size_t)(hh * 16) * CTX_ + t0, CTX_);
      acc[hh] = WMMA_BF16(pa, vb, acc[hh]);
    }
  }

  // ---- normalize and scatter to (b, s, h*HD + hd) f32
#pragma unroll
  for (int j = 0; j < 8; ++j) {
    const float inv = 1.0f / l[j];
    const size_t row = (size_t)(b * S_ + qt * 16 + rb + j) * E_ + (size_t)h * HD_;
#pragma unroll
    for (int hh = 0; hh < 4; ++hh)
      attnf[row + hh * 16 + n] = acc[hh][j] * inv;
  }
}

// ---------------------------------------------------------------- launch
extern "C" void kernel_launch(void* const* d_in, const int* in_sizes, int n_in,
                              void* d_out, int out_size, void* d_ws, size_t ws_size,
                              hipStream_t stream) {
  const float* x       = (const float*)d_in[0];
  const float* cosb    = (const float*)d_in[1];
  const float* sinb    = (const float*)d_in[2];
  // d_in[3] (mask) handled analytically; d_in[10] (start_pos) is constant P_.
  const float* cache_k = (const float*)d_in[4];
  const float* cache_v = (const float*)d_in[5];
  const float* Wq      = (const float*)d_in[6];
  const float* Wk      = (const float*)d_in[7];
  const float* Wv      = (const float*)d_in[8];
  const float* Wo      = (const float*)d_in[9];
  float* out = (float*)d_out;

  char* ws = (char*)d_ws;
  const size_t MB = 1024ull * 1024ull;
  bf16*  xb    = (bf16*)(ws + 0);        //  8MB bf16 x          (reused as attnb)
  bf16*  wqb   = (bf16*)(ws + 8  * MB);  //  8MB
  bf16*  wkb   = (bf16*)(ws + 16 * MB);  //  2MB
  bf16*  wvb   = (bf16*)(ws + 18 * MB);  //  2MB
  bf16*  wob   = (bf16*)(ws + 20 * MB);  //  8MB
  float* qf    = (float*)(ws + 28 * MB); // 16MB f32 q pre-rope  (reused as attnf)
  float* kf    = (float*)(ws + 44 * MB); //  4MB
  float* vf    = (float*)(ws + 48 * MB); //  4MB
  bf16*  qbb   = (bf16*)(ws + 52 * MB);  //  8MB (B,H,S,HD) roped q
  bf16*  keysb = (bf16*)(ws + 60 * MB);  //  4MB (B,KV,CTX,HD)
  bf16*  valst = (bf16*)(ws + 64 * MB);  //  4MB (B,KV,HD,CTX)
  float* attnf = qf;                     // qf dead after rope
  bf16*  attnb = xb;                     // xb dead after projections

  const int TB = 256;
  auto cdiv = [](int a, int b) { return (a + b - 1) / b; };

  // 1) bf16 operand staging
  k_f32_to_bf16<<<cdiv(B_ * S_ * E_, TB), TB, 0, stream>>>(x,  xb,  B_ * S_ * E_);
  k_f32_to_bf16<<<cdiv(E_ * E_,   TB), TB, 0, stream>>>(Wq, wqb, E_ * E_);
  k_f32_to_bf16<<<cdiv(KVD_ * E_, TB), TB, 0, stream>>>(Wk, wkb, KVD_ * E_);
  k_f32_to_bf16<<<cdiv(KVD_ * E_, TB), TB, 0, stream>>>(Wv, wvb, KVD_ * E_);
  k_f32_to_bf16<<<cdiv(E_ * E_,   TB), TB, 0, stream>>>(Wo, wob, E_ * E_);

  // 2) QKV projections (WMMA), 128x128 per block / 64x64 per wave
  dim3 blk(128);
  dim3 gq(E_ / 128,   (B_ * S_) / 128);
  dim3 gk(KVD_ / 128, (B_ * S_) / 128);
  k_gemm_bf16_nt<<<gq, blk, 0, stream>>>(xb, wqb, qf, B_ * S_, E_,   E_);
  k_gemm_bf16_nt<<<gk, blk, 0, stream>>>(xb, wkb, kf, B_ * S_, KVD_, E_);
  k_gemm_bf16_nt<<<gk, blk, 0, stream>>>(xb, wvb, vf, B_ * S_, KVD_, E_);

  // 3) RoPE + KV cache staging (bf16, V transposed for PV fragment loads)
  k_rope_q     <<<cdiv(B_ * H_ * S_ * HD_,   TB), TB, 0, stream>>>(qf, cosb, sinb, qbb);
  k_build_keys <<<cdiv(B_ * KV_ * CTX_ * HD_, TB), TB, 0, stream>>>(kf, cache_k, cosb, sinb, keysb);
  k_build_valsT<<<cdiv(B_ * KV_ * HD_ * CTX_, TB), TB, 0, stream>>>(vf, cache_v, valst);

  // 4) flash attention (WMMA scores + WMMA PV, online softmax)
  dim3 ga(S_ / 16, H_, B_);
  k_flash_attn<<<ga, dim3(32), 0, stream>>>(qbb, keysb, valst, attnf);

  // 5) output projection (WMMA) -> d_out f32
  k_f32_to_bf16<<<cdiv(B_ * S_ * E_, TB), TB, 0, stream>>>(attnf, attnb, B_ * S_ * E_);
  k_gemm_bf16_nt<<<gq, blk, 0, stream>>>(attnb, wob, out, B_ * S_, E_, E_);

  (void)in_sizes; (void)n_in; (void)out_size; (void)ws_size;
}
